// Mamba2Block_76390288326770
// MI455X (gfx1250) — matrix operand order repaired and checked by reading, not compiled, not run on hardware
//
#include <hip/hip_runtime.h>
#include <hip/hip_bf16.h>
#include <stdint.h>

// ---------------- problem constants (match reference) ----------------
#define BATCH    2
#define SEQ      2048
#define DMODEL   1024
#define DSTATE   128
#define DCONV    4
#define DINNER   2048
#define NH       32
#define HD       64
#define CHUNK    64
#define NC       (SEQ / CHUNK)            // 32
#define CONVDIM  (DINNER + 2 * DSTATE)    // 2304
#define DPROJ    (2 * DINNER + 2 * DSTATE + NH)  // 4384
#define MROWS    (BATCH * SEQ)            // 4096

typedef _Float16 half8  __attribute__((ext_vector_type(8)));
typedef _Float16 v16h   __attribute__((ext_vector_type(16)));
typedef float    v8f    __attribute__((ext_vector_type(8)));

// ---------------- WMMA fragment helpers (NT: A[M,K], B[N,K], K contiguous) ----------------
__device__ __forceinline__ v16h load_afrag(const _Float16* p) {
    // lane (lr, lg) holds row lr, k {16*lg..16*lg+15} contiguous
    half8 alo = *(const half8*)(p);
    half8 ahi = *(const half8*)(p + 8);
    v16h a;
#pragma unroll
    for (int i = 0; i < 8; ++i) { a[i] = alo[i]; a[8 + i] = ahi[i]; }
    return a;
}

// one 16x16 output tile, accumulate over K
__device__ __forceinline__ v8f wmma16_nt(const _Float16* __restrict__ A, int lda,
                                         const _Float16* __restrict__ B, int ldb,
                                         int K, v8f acc) {
    const int lane = threadIdx.x & 31;
    const int lr = lane & 15, lg = lane >> 4;
    const _Float16* ar = A + (size_t)lr * lda + lg * 16;
    const _Float16* br = B + (size_t)lr * ldb + lg * 16;
    for (int k0 = 0; k0 < K; k0 += 32) {
        v16h a  = load_afrag(ar + k0);
        v16h bv = *(const v16h*)(br + k0);
        acc = __builtin_amdgcn_wmma_f32_16x16x32_f16(false, a, false, bv,
                                                     (short)0, acc, false, false);
    }
    return acc;
}

__device__ __forceinline__ void wmma_store16(float* __restrict__ C, int ldc, v8f c) {
    const int lane = threadIdx.x & 31;
    const int lr = lane & 15, lg = lane >> 4;
    float* cr = C + (size_t)(lg * 8) * ldc + lr;
#pragma unroll
    for (int r = 0; r < 8; ++r) cr[(size_t)r * ldc] = c[r];
}

// ---------------- fp32 -> f16 convert ----------------
__global__ void convert_f32_f16(const float* __restrict__ src,
                                _Float16* __restrict__ dst, int n) {
    for (int i = blockIdx.x * blockDim.x + threadIdx.x; i < n;
         i += gridDim.x * blockDim.x)
        dst[i] = (_Float16)src[i];
}

// ---------------- big WMMA NT GEMM: C[M,N] = A[M,K] * B[N,K]^T ----------------
// Wave tile 32x32 (4 accumulators); block 256 threads = 8 waves -> 256x32 tile.
// M % 256 == 0, N % 32 == 0, K % 32 == 0. Operands are L2-resident (<=18 MB).
__global__ void wmma_gemm_nt(const _Float16* __restrict__ A,
                             const _Float16* __restrict__ Bw,
                             float* __restrict__ C,
                             int M, int N, int K) {
    const int lane = threadIdx.x & 31;
    const int wave = threadIdx.x >> 5;
    const int m0 = blockIdx.x * 256 + wave * 32;
    const int n0 = blockIdx.y * 32;
    const int lr = lane & 15, lg = lane >> 4;

    v8f c00 = {}, c01 = {}, c10 = {}, c11 = {};
    const _Float16* a0 = A + (size_t)(m0 + lr) * K + lg * 16;
    const _Float16* a1 = a0 + (size_t)16 * K;
    const _Float16* b0 = Bw + (size_t)(n0 + lr) * K + lg * 16;
    const _Float16* b1 = b0 + (size_t)16 * K;

    for (int k0 = 0; k0 < K; k0 += 32) {
        v16h av0 = load_afrag(a0 + k0);
        v16h av1 = load_afrag(a1 + k0);
        v16h bv0 = *(const v16h*)(b0 + k0);
        v16h bv1 = *(const v16h*)(b1 + k0);
        c00 = __builtin_amdgcn_wmma_f32_16x16x32_f16(false, av0, false, bv0, (short)0, c00, false, false);
        c01 = __builtin_amdgcn_wmma_f32_16x16x32_f16(false, av0, false, bv1, (short)0, c01, false, false);
        c10 = __builtin_amdgcn_wmma_f32_16x16x32_f16(false, av1, false, bv0, (short)0, c10, false, false);
        c11 = __builtin_amdgcn_wmma_f32_16x16x32_f16(false, av1, false, bv1, (short)0, c11, false, false);
    }
    float* cr0 = C + (size_t)(m0 + lg * 8) * N + n0 + lr;
    float* cr1 = cr0 + (size_t)16 * N;
#pragma unroll
    for (int r = 0; r < 8; ++r) {
        cr0[(size_t)r * N]      = c00[r];
        cr0[(size_t)r * N + 16] = c01[r];
        cr1[(size_t)r * N]      = c10[r];
        cr1[(size_t)r * N + 16] = c11[r];
    }
}

// ---------------- depthwise conv4 + SiLU + split (B/C straight to f16) ----------------
__global__ void conv_silu_kernel(const float* __restrict__ zx,
                                 const float* __restrict__ conv_w,
                                 const float* __restrict__ conv_b,
                                 float* __restrict__ xs,
                                 _Float16* __restrict__ Bm16,
                                 _Float16* __restrict__ Cm16) {
    const int total = MROWS * CONVDIM;
    for (int i = blockIdx.x * blockDim.x + threadIdx.x; i < total;
         i += gridDim.x * blockDim.x) {
        int c  = i % CONVDIM;
        int bl = i / CONVDIM;
        int b  = bl / SEQ;
        int l  = bl % SEQ;
        float acc = conv_b[c];
#pragma unroll
        for (int j = 0; j < DCONV; ++j) {
            int lj = l - (DCONV - 1) + j;
            if (lj >= 0)
                acc += zx[(size_t)(b * SEQ + lj) * DPROJ + DINNER + c] *
                       conv_w[c * DCONV + j];
        }
        float v = acc * (1.f / (1.f + __expf(-acc)));  // silu
        if (c < DINNER)
            xs[(size_t)bl * DINNER + c] = v;
        else if (c < DINNER + DSTATE)
            Bm16[(size_t)bl * DSTATE + (c - DINNER)] = (_Float16)v;
        else
            Cm16[(size_t)bl * DSTATE + (c - DINNER - DSTATE)] = (_Float16)v;
    }
}

// ---------------- dt = softplus(raw + bias); a = dt * (-exp(A_log)) ----------------
__global__ void dt_kernel(const float* __restrict__ zx,
                          const float* __restrict__ dt_bias,
                          const float* __restrict__ A_log,
                          float* __restrict__ dtb,
                          float* __restrict__ ab) {
    const int total = MROWS * NH;
    for (int i = blockIdx.x * blockDim.x + threadIdx.x; i < total;
         i += gridDim.x * blockDim.x) {
        int h  = i % NH;
        int bl = i / NH;
        float x = zx[(size_t)bl * DPROJ + DINNER + CONVDIM + h] + dt_bias[h];
        float sp = (x > 20.f) ? x : log1pf(__expf(x));
        dtb[i] = sp;
        ab[i]  = -__expf(A_log[h]) * sp;
    }
}

// ---------------- per-chunk B transpose: BmT16[bc][n][z] ----------------
__global__ void bmt_kernel(const _Float16* __restrict__ Bm16,
                           _Float16* __restrict__ BmT16) {
    int bc = blockIdx.x;
    int b = bc / NC, c = bc % NC;
    int rowbase = b * SEQ + c * CHUNK;
    for (int t = threadIdx.x; t < CHUNK * DSTATE; t += blockDim.x) {
        int n = t >> 6, z = t & 63;
        BmT16[(size_t)bc * DSTATE * CHUNK + t] =
            Bm16[(size_t)(rowbase + z) * DSTATE + n];
    }
}

// ---------------- G[b,c][s][z] = sum_n C[s,n] B[z,n]   (WMMA, K=128) ----------------
__global__ void g_gemm_kernel(const _Float16* __restrict__ Cm16,
                              const _Float16* __restrict__ Bm16,
                              float* __restrict__ G) {
    int bc = blockIdx.x;
    int b = bc / NC, c = bc % NC;
    int rowbase = b * SEQ + c * CHUNK;
    int wave = threadIdx.x >> 5;
    for (int t = wave; t < 16; t += 8) {      // 4x4 tiles of 16x16
        int mi = t >> 2, ni = t & 3;
        v8f acc = {};
        acc = wmma16_nt(Cm16 + (size_t)(rowbase + mi * 16) * DSTATE, DSTATE,
                        Bm16 + (size_t)(rowbase + ni * 16) * DSTATE, DSTATE,
                        DSTATE, acc);
        wmma_store16(G + (size_t)bc * CHUNK * CHUNK + mi * 16 * CHUNK + ni * 16,
                     CHUNK, acc);
    }
}

// ---------------- per (b,chunk,head) prep: cumsum, decays, W=G*L (f16),
//                  xd^T (f16) and decay-folded xd^T (f16).
//                  G tile staged LDS via async global->LDS loads. ----------------
__global__ void chunk_prep_kernel(const float* __restrict__ xs,
                                  const float* __restrict__ dtb,
                                  const float* __restrict__ ab,
                                  const float* __restrict__ G,
                                  _Float16* __restrict__ Wmat16,
                                  _Float16* __restrict__ xdT16,
                                  _Float16* __restrict__ xdecT16,
                                  float* __restrict__ acumbuf,
                                  float* __restrict__ asum) {
    __shared__ __attribute__((aligned(16))) float lG[CHUNK * CHUNK];
    __shared__ __attribute__((aligned(16))) float lxd[CHUNK * HD];
    __shared__ float lacum[CHUNK];
    __shared__ float ldec[CHUNK];

    int bcn = blockIdx.x;
    int h = bcn % NH, bc = bcn / NH;
    int b = bc / NC, c = bc % NC;
    int rowbase = b * SEQ + c * CHUNK;
    int tid = threadIdx.x;

    // --- async-stage G chunk (16 KB) into LDS: ASYNCcnt-tracked b128 copies ---
    // (low 32 bits of a flat shared pointer are the wave-relative LDS address)
#pragma unroll
    for (int it = 0; it < (CHUNK * CHUNK / 4) / 256; ++it) {
        int i = tid + it * 256;
        uint32_t lp = (uint32_t)(uintptr_t)&lG[i * 4];
        uint64_t gp = (uint64_t)(uintptr_t)(G + (size_t)bc * CHUNK * CHUNK + i * 4);
        asm volatile("global_load_async_to_lds_b128 %0, %1, off"
                     :: "v"(lp), "v"(gp) : "memory");
    }

    // --- cumulative decay within the chunk ---
    if (tid < CHUNK) lacum[tid] = ab[(size_t)(rowbase + tid) * NH + h];
    __syncthreads();
    if (tid == 0) {
        float run = 0.f;
        for (int z = 0; z < CHUNK; ++z) { run += lacum[z]; lacum[z] = run; }
    }
    __syncthreads();
    float aend = lacum[CHUNK - 1];
    if (tid < CHUNK) {
        acumbuf[(size_t)bcn * CHUNK + tid] = lacum[tid];
        ldec[tid] = __expf(aend - lacum[tid]);
    }
    if (tid == 0) asum[(b * NH + h) * NC + c] = aend;

    // --- xd[z][p] = xs * dt ---
    for (int t = tid; t < CHUNK * HD; t += blockDim.x) {
        int z = t >> 6, p = t & 63;
        lxd[t] = xs[(size_t)(rowbase + z) * DINNER + h * HD + p] *
                 dtb[(size_t)(rowbase + z) * NH + h];
    }

    // --- wait for async G staging, then sync whole workgroup ---
    asm volatile("s_wait_asynccnt 0x0" ::: "memory");
    __syncthreads();

    // --- W[s][z] = masked G * exp(acum_s - acum_z), f16 (A operand, K=z contiguous) ---
    for (int t = tid; t < CHUNK * CHUNK; t += blockDim.x) {
        int s = t >> 6, z = t & 63;
        float w = (z <= s) ? lG[t] * __expf(lacum[s] - lacum[z]) : 0.f;
        Wmat16[(size_t)bcn * CHUNK * CHUNK + t] = (_Float16)w;
    }
    // --- xdT[p][z] and decay-folded xdecT[p][z], f16 ---
    for (int t = tid; t < CHUNK * HD; t += blockDim.x) {
        int p = t >> 6, z = t & 63;
        float v = lxd[z * HD + p];
        xdT16[(size_t)bcn * CHUNK * HD + t]   = (_Float16)v;
        xdecT16[(size_t)bcn * CHUNK * HD + t] = (_Float16)(ldec[z] * v);
    }
}

// ---------------- Y_diag[s][p] = W @ xd  (WMMA, K=64) ----------------
__global__ void ydiag_gemm_kernel(const _Float16* __restrict__ Wmat16,
                                  const _Float16* __restrict__ xdT16,
                                  float* __restrict__ ybuf) {
    int bcn = blockIdx.x;
    int h = bcn % NH, bc = bcn / NH;
    int b = bc / NC, c = bc % NC;
    int rowbase = b * SEQ + c * CHUNK;
    const _Float16* A  = Wmat16 + (size_t)bcn * CHUNK * CHUNK;  // [s][z]
    const _Float16* Bp = xdT16  + (size_t)bcn * CHUNK * HD;     // [p][z]
    int wave = threadIdx.x >> 5;
    for (int t = wave; t < 16; t += 8) {
        int mi = t >> 2, ni = t & 3;
        v8f acc = {};
        acc = wmma16_nt(A + mi * 16 * CHUNK, CHUNK,
                        Bp + ni * 16 * CHUNK, CHUNK, CHUNK, acc);
        wmma_store16(ybuf + (size_t)(rowbase + mi * 16) * DINNER + h * HD + ni * 16,
                     DINNER, acc);
    }
}

// ---------------- states[p][n] = sum_z decay[z]*xd[z,p]*B[z,n]  (WMMA, K=64) ----------------
__global__ void states_gemm_kernel(const _Float16* __restrict__ xdecT16,
                                   const _Float16* __restrict__ BmT16,
                                   float* __restrict__ states) {
    int bcn = blockIdx.x;
    int bc = bcn / NH;
    const _Float16* A  = xdecT16 + (size_t)bcn * CHUNK * HD;     // [p][z]
    const _Float16* Bp = BmT16   + (size_t)bc * DSTATE * CHUNK;  // [n][z]
    int wave = threadIdx.x >> 5;
    for (int t = wave; t < 32; t += 8) {      // 4 (p) x 8 (n) tiles
        int mi = t >> 3, ni = t & 7;
        v8f acc = {};
        acc = wmma16_nt(A + mi * 16 * CHUNK, CHUNK,
                        Bp + ni * 16 * CHUNK, CHUNK, CHUNK, acc);
        wmma_store16(states + (size_t)bcn * HD * DSTATE + mi * 16 * DSTATE + ni * 16,
                     DSTATE, acc);
    }
}

// ---------------- inter-chunk sequential scan (in-place -> state entering chunk) ----------------
__global__ void ssd_scan_kernel(float* __restrict__ states,
                                const float* __restrict__ asum) {
    int bh = blockIdx.x;
    int b = bh / NH, h = bh % NH;
    float run[32];
#pragma unroll
    for (int j = 0; j < 32; ++j) run[j] = 0.f;
    for (int c = 0; c < NC; ++c) {
        float dec = __expf(asum[bh * NC + c]);
        size_t base = ((size_t)((b * NC + c) * NH + h)) * (HD * DSTATE);
#pragma unroll
        for (int j = 0; j < 32; ++j) {
            size_t idx = base + threadIdx.x + j * 256;
            float tmp = states[idx];
            states[idx] = run[j];
            run[j] = run[j] * dec + tmp;
        }
    }
}

// ---------------- Y_off (WMMA, K=128) with fused exp-scale + D*x epilogue ----------------
__global__ void yoff_gemm_kernel(const _Float16* __restrict__ Cm16,
                                 const _Float16* __restrict__ states16,
                                 const float* __restrict__ acumbuf,
                                 const float* __restrict__ xs,
                                 const float* __restrict__ Dp,
                                 float* __restrict__ ybuf) {
    int bcn = blockIdx.x;
    int h = bcn % NH, bc = bcn / NH;
    int b = bc / NC, c = bc % NC;
    int rowbase = b * SEQ + c * CHUNK;
    const _Float16* A  = Cm16    + (size_t)rowbase * DSTATE;     // [s][n]
    const _Float16* Bp = states16 + (size_t)bcn * HD * DSTATE;   // [p][n]
    float dh = Dp[h];
    int wave = threadIdx.x >> 5;
    const int lane = threadIdx.x & 31;
    const int lr = lane & 15, lg = lane >> 4;
    for (int t = wave; t < 16; t += 8) {
        int mi = t >> 2, ni = t & 3;
        v8f acc = {};
        acc = wmma16_nt(A + mi * 16 * DSTATE, DSTATE,
                        Bp + ni * 16 * DSTATE, DSTATE, DSTATE, acc);
        int s0 = mi * 16 + lg * 8;
        int p  = ni * 16 + lr;
#pragma unroll
        for (int r = 0; r < 8; ++r) {
            int s = s0 + r;
            float esc = __expf(acumbuf[(size_t)bcn * CHUNK + s]);
            size_t yi = (size_t)(rowbase + s) * DINNER + h * HD + p;
            ybuf[yi] += esc * acc[r] + dh * xs[yi];
        }
    }
}

// ---------------- gating + RMSNorm -> f16 ----------------
__global__ void gate_norm_kernel(const float* __restrict__ zx,
                                 const float* __restrict__ normw,
                                 const float* __restrict__ ybuf,
                                 _Float16* __restrict__ y16) {
    __shared__ float red[256];
    int row = blockIdx.x;
    float vals[8];
    float ss = 0.f;
#pragma unroll
    for (int j = 0; j < 8; ++j) {
        int e = threadIdx.x + j * 256;
        float zv = zx[(size_t)row * DPROJ + e];
        float yv = ybuf[(size_t)row * DINNER + e] *
                   (zv * (1.f / (1.f + __expf(-zv))));
        vals[j] = yv;
        ss += yv * yv;
    }
    red[threadIdx.x] = ss;
    __syncthreads();
    for (int off = 128; off > 0; off >>= 1) {
        if (threadIdx.x < off) red[threadIdx.x] += red[threadIdx.x + off];
        __syncthreads();
    }
    float scale = rsqrtf(red[0] / (float)DINNER + 1e-5f);
#pragma unroll
    for (int j = 0; j < 8; ++j) {
        int e = threadIdx.x + j * 256;
        y16[(size_t)row * DINNER + e] = (_Float16)(vals[j] * scale * normw[e]);
    }
}

// ---------------- host side ----------------
extern "C" void kernel_launch(void* const* d_in, const int* in_sizes, int n_in,
                              void* d_out, int out_size, void* d_ws, size_t ws_size,
                              hipStream_t stream) {
    const float* x      = (const float*)d_in[0];
    const float* W_in   = (const float*)d_in[1];
    const float* conv_w = (const float*)d_in[2];
    const float* conv_b = (const float*)d_in[3];
    const float* dtbias = (const float*)d_in[4];
    const float* A_log  = (const float*)d_in[5];
    const float* D_par  = (const float*)d_in[6];
    const float* norm_w = (const float*)d_in[7];
    const float* W_out  = (const float*)d_in[8];
    float* out = (float*)d_out;

    char* p = (char*)d_ws;
    auto alloc = [&](size_t bytes) -> void* {
        void* r = (void*)p;
        p += (bytes + 255) & ~(size_t)255;
        return r;
    };
    _Float16* x16     = (_Float16*)alloc((size_t)MROWS * DMODEL * 2);
    _Float16* Win16   = (_Float16*)alloc((size_t)DPROJ * DMODEL * 2);
    _Float16* Wout16  = (_Float16*)alloc((size_t)DMODEL * DINNER * 2);
    float*    zx      = (float*)alloc((size_t)MROWS * DPROJ * 4);
    float*    xs      = (float*)alloc((size_t)MROWS * DINNER * 4);
    _Float16* Bm16    = (_Float16*)alloc((size_t)MROWS * DSTATE * 2);
    _Float16* Cm16    = (_Float16*)alloc((size_t)MROWS * DSTATE * 2);
    _Float16* BmT16   = (_Float16*)alloc((size_t)BATCH * NC * DSTATE * CHUNK * 2);
    float*    dtb     = (float*)alloc((size_t)MROWS * NH * 4);
    float*    ab      = (float*)alloc((size_t)MROWS * NH * 4);
    float*    G       = (float*)alloc((size_t)BATCH * NC * CHUNK * CHUNK * 4);
    _Float16* Wmat16  = (_Float16*)alloc((size_t)BATCH * NC * NH * CHUNK * CHUNK * 2);
    _Float16* xdT16   = (_Float16*)alloc((size_t)BATCH * NC * NH * CHUNK * HD * 2);
    _Float16* xdecT16 = (_Float16*)alloc((size_t)BATCH * NC * NH * CHUNK * HD * 2);
    float*    states  = (float*)alloc((size_t)BATCH * NC * NH * HD * DSTATE * 4);
    _Float16* states16= (_Float16*)alloc((size_t)BATCH * NC * NH * HD * DSTATE * 2);
    float*    acumb   = (float*)alloc((size_t)BATCH * NC * NH * CHUNK * 4);
    float*    asum    = (float*)alloc((size_t)BATCH * NH * NC * 4);
    float*    ybuf    = (float*)alloc((size_t)MROWS * DINNER * 4);
    _Float16* y16     = (_Float16*)alloc((size_t)MROWS * DINNER * 2);

    // 1) converts
    convert_f32_f16<<<2048, 256, 0, stream>>>(x, x16, MROWS * DMODEL);
    convert_f32_f16<<<2048, 256, 0, stream>>>(W_in, Win16, DPROJ * DMODEL);
    convert_f32_f16<<<2048, 256, 0, stream>>>(W_out, Wout16, DMODEL * DINNER);

    // 2) in_proj (WMMA): zx[M, DPROJ] = x16 * Win16^T
    wmma_gemm_nt<<<dim3(MROWS / 256, DPROJ / 32), 256, 0, stream>>>(
        x16, Win16, zx, MROWS, DPROJ, DMODEL);

    // 3) conv + silu + split ; dt/a
    conv_silu_kernel<<<4096, 256, 0, stream>>>(zx, conv_w, conv_b, xs, Bm16, Cm16);
    dt_kernel<<<512, 256, 0, stream>>>(zx, dtbias, A_log, dtb, ab);

    // 4) SSD — all matmuls on WMMA
    bmt_kernel<<<BATCH * NC, 256, 0, stream>>>(Bm16, BmT16);
    g_gemm_kernel<<<BATCH * NC, 256, 0, stream>>>(Cm16, Bm16, G);
    chunk_prep_kernel<<<BATCH * NC * NH, 256, 0, stream>>>(
        xs, dtb, ab, G, Wmat16, xdT16, xdecT16, acumb, asum);
    ydiag_gemm_kernel<<<BATCH * NC * NH, 256, 0, stream>>>(Wmat16, xdT16, ybuf);
    states_gemm_kernel<<<BATCH * NC * NH, 256, 0, stream>>>(xdecT16, BmT16, states);
    ssd_scan_kernel<<<BATCH * NH, 256, 0, stream>>>(states, asum);
    convert_f32_f16<<<2048, 256, 0, stream>>>(
        states, states16, BATCH * NC * NH * HD * DSTATE);
    yoff_gemm_kernel<<<BATCH * NC * NH, 256, 0, stream>>>(
        Cm16, states16, acumb, xs, D_par, ybuf);

    // 5) gating + RMSNorm -> f16
    gate_norm_kernel<<<MROWS, 256, 0, stream>>>(zx, norm_w, ybuf, y16);

    // 6) out_proj (WMMA): out[M, DMODEL] = y16 * Wout16^T
    wmma_gemm_nt<<<dim3(MROWS / 256, DMODEL / 32), 256, 0, stream>>>(
        y16, Wout16, out, MROWS, DMODEL, DINNER);
}
